// ChunkedCrossEntropyLoss_49804440765055
// MI455X (gfx1250) — compile-verified, matrix operand order checked by reading, not run on hardware
//
#include <hip/hip_runtime.h>

// ---------------------------------------------------------------------------
// Chunked cross-entropy loss for MI455X (gfx1250, wave32, WMMA).
//   hidden_states: [4,2048,2048] f32 -> N=8192 rows, D=2048
//   targets:       [8192] int64
//   weight_matrix: [131072,2048] f32  (V=131072)
//   out: scalar f32 loss = mean(logsumexp(h@W^T) - h.W[tgt])
//
// Memory-bound on W (1 GiB @ 23.3 TB/s ~= 46us). Strategy:
//   k0: h f32 -> bf16 (32 MiB, stays hot in 192MB L2)
//   k1: target logits via gather + wave dot product
//   k2: streaming bf16-WMMA GEMM, double-buffered LDS; A tile staged with
//       CDNA5 async global->LDS (ASYNCcnt), B tile register-staged with
//       in-flight f32->bf16; register-resident online logsumexp.
//   k3: merge 32 partial (m,s) per row, subtract target logit, mean.
// ---------------------------------------------------------------------------

#define DDIM   2048
#define NROWS  8192
#define VOCAB  131072
#define VSPLIT 32
#define CPS    (VOCAB / VSPLIT)   // 4096 cols per split
#define MTILE  128                // rows per block
#define NTILE  128                // cols per inner tile
#define KSTEP  32                 // bf16 WMMA K

typedef __attribute__((ext_vector_type(16))) __bf16 v16bf;
typedef __attribute__((ext_vector_type(8)))  __bf16 v8bf;
typedef __attribute__((ext_vector_type(8)))  float  v8f;

union Frag { v16bf v; v8bf h[2]; };

// f32x2 -> packed bf16x2, round-half-up + v_perm_b32 byte pack (3 VALU).
__device__ __forceinline__ unsigned int pk_bf16(float a, float b) {
  unsigned int ua = __float_as_uint(a) + 0x8000u;
  unsigned int ub = __float_as_uint(b) + 0x8000u;
  return __builtin_amdgcn_perm(ub, ua, 0x07060302u);  // {hi(b),hi(a)}
}

// CDNA5 async global->LDS, 16B per lane (GLOBAL_LOAD_ASYNC_TO_LDS_B128,
// tracked by ASYNCcnt). lds_off = low 32 bits of a generic shared pointer.
__device__ __forceinline__ void async_g2l_b128(unsigned lds_off, const void* g) {
  unsigned long long ga = (unsigned long long)g;
  asm volatile("global_load_async_to_lds_b128 %0, %1, off"
               :: "v"(lds_off), "v"(ga) : "memory");
}
__device__ __forceinline__ void wait_asynccnt0() {
  asm volatile("s_wait_asynccnt 0x0" ::: "memory");
}

// ---------------- k0: h f32 -> bf16 ----------------------------------------
__global__ __launch_bounds__(256)
void cce_convert_h(const float* __restrict__ h, unsigned short* __restrict__ hb) {
  const long long base = ((long long)blockIdx.x * 256 + threadIdx.x) * 8;
  if (base >= (long long)NROWS * DDIM) return;
  float4 f0 = *(const float4*)(h + base);
  float4 f1 = *(const float4*)(h + base + 4);
  uint4 o;
  o.x = pk_bf16(f0.x, f0.y);
  o.y = pk_bf16(f0.z, f0.w);
  o.z = pk_bf16(f1.x, f1.y);
  o.w = pk_bf16(f1.z, f1.w);
  *(uint4*)(hb + base) = o;
}

// ---------------- k1: target logits (one wave per row) ---------------------
__global__ __launch_bounds__(256)
void cce_target_logits(const float* __restrict__ h,
                       const long long* __restrict__ tgt,   // int64 per reference
                       const float* __restrict__ W,
                       float* __restrict__ tl) {
  const int row  = blockIdx.x * 8 + (threadIdx.x >> 5);
  const int lane = threadIdx.x & 31;
  if (row >= NROWS) return;
  const long long t = tgt[row];
  const float4* hr = (const float4*)(h + (long long)row * DDIM);
  const float4* wr = (const float4*)(W + t * (long long)DDIM);
  float s = 0.f;
  #pragma unroll 4
  for (int k = lane; k < DDIM / 4; k += 32) {
    float4 a = hr[k], b = wr[k];
    s += a.x * b.x + a.y * b.y + a.z * b.z + a.w * b.w;
  }
  #pragma unroll
  for (int off = 1; off < 32; off <<= 1) s += __shfl_xor(s, off, 32);
  if (lane == 0) tl[row] = s;
}

// ---------------- k2: WMMA GEMM + online logsumexp -------------------------
__global__ __launch_bounds__(256)
void cce_lse_partial(const unsigned short* __restrict__ hb,  // [NROWS][D] bf16
                     const float* __restrict__ W,            // [VOCAB][D] f32
                     float* __restrict__ pm,                 // [NROWS][VSPLIT]
                     float* __restrict__ ps) {
  __shared__ __align__(16) unsigned short As[2][MTILE * KSTEP];  // 2 x 8 KB
  __shared__ __align__(16) unsigned short Bs[2][NTILE * KSTEP];  // 2 x 8 KB

  const int tid  = threadIdx.x;
  const int wave = tid >> 5;
  const int lane = tid & 31;
  const int grp  = lane >> 4;        // 0: rows r..r+7 ; 1: rows r+8..
  const int l16  = lane & 15;

  const int row_base = blockIdx.x * MTILE;
  const int col_base = blockIdx.y * CPS;
  const int wrow     = row_base + wave * 16;

  const int srow = tid >> 1;                 // staging row/col
  const int koff = (tid & 1) * 16;           // staging k-offset (halves)
  const unsigned short* aptr = hb + (long long)(row_base + srow) * DDIM + koff;
  const unsigned aoff0 =
      (unsigned)(unsigned long long)&As[0][srow * KSTEP + koff];
  const unsigned aoff1 =
      (unsigned)(unsigned long long)&As[1][srow * KSTEP + koff];

  float m_run[8], s_run[8];
  #pragma unroll
  for (int r = 0; r < 8; ++r) { m_run[r] = -1e30f; s_run[r] = 0.f; }

  for (int ct = 0; ct < CPS / NTILE; ++ct) {
    const float* bptr =
        W + (long long)(col_base + ct * NTILE + srow) * DDIM + koff;
    v8f acc[8] = {};
    float4 f0, f1, f2, f3;          // B stage (f32, convert at LDS-store time)

    auto stage_a = [&](int kk, unsigned aoff) {   // async global->LDS, 32 B
      async_g2l_b128(aoff,      aptr + kk);
      async_g2l_b128(aoff + 16, aptr + kk + 8);
    };
    auto load_b = [&](int kk) {
      f0 = *(const float4*)(bptr + kk);
      f1 = *(const float4*)(bptr + kk + 4);
      f2 = *(const float4*)(bptr + kk + 8);
      f3 = *(const float4*)(bptr + kk + 12);
    };
    auto store_b = [&](int buf) {
      uint4 p0, p1;
      p0.x = pk_bf16(f0.x, f0.y); p0.y = pk_bf16(f0.z, f0.w);
      p0.z = pk_bf16(f1.x, f1.y); p0.w = pk_bf16(f1.z, f1.w);
      p1.x = pk_bf16(f2.x, f2.y); p1.y = pk_bf16(f2.z, f2.w);
      p1.z = pk_bf16(f3.x, f3.y); p1.w = pk_bf16(f3.z, f3.w);
      *(uint4*)&Bs[buf][srow * KSTEP + koff]     = p0;
      *(uint4*)&Bs[buf][srow * KSTEP + koff + 8] = p1;
    };

    stage_a(0, aoff0);              // prologue -> buffer 0
    load_b(0);
    store_b(0);
    int cur = 0;

    for (int kk = 0; kk < DDIM; kk += KSTEP) {
      wait_asynccnt0();                      // our async writes to buf cur done
      __syncthreads();                       // everyone's writes visible
      const bool more = (kk + KSTEP) < DDIM;
      if (more) {
        stage_a(kk + KSTEP, cur ? aoff0 : aoff1);  // async into other buffer
        load_b(kk + KSTEP);                        // overlap with WMMA below
      }

      // Preload ALL fragments, then 8 back-to-back WMMAs (sched groups below).
      // A frag (ISA 16-bit 16x32): lane<16 K{0..7,16..23}, lane>=16 +8.
      const int kb = grp * 8;
      Frag a;
      a.h[0] = *(const v8bf*)&As[cur][(wave * 16 + l16) * KSTEP + kb];
      a.h[1] = *(const v8bf*)&As[cur][(wave * 16 + l16) * KSTEP + kb + 16];
      Frag b[8];
      #pragma unroll
      for (int c2 = 0; c2 < 8; ++c2) {
        b[c2].h[0] = *(const v8bf*)&Bs[cur][(c2 * 16 + l16) * KSTEP + kb];
        b[c2].h[1] = *(const v8bf*)&Bs[cur][(c2 * 16 + l16) * KSTEP + kb + 16];
      }
      #pragma unroll
      for (int c2 = 0; c2 < 8; ++c2)
        acc[c2] = __builtin_amdgcn_wmma_f32_16x16x32_bf16(
            false, a.v, false, b[c2].v, (short)0, acc[c2], false, false);
      __builtin_amdgcn_sched_group_barrier(0x100, 18, 0);  // 18 DS reads first
      __builtin_amdgcn_sched_group_barrier(0x008, 8, 0);   // then 8 matrix ops

      if (more) store_b(cur ^ 1);            // write other buffer; no race
      cur ^= 1;
    }

    // ---- online LSE update over this 128-col tile (wave-local) ----
    #pragma unroll
    for (int r = 0; r < 8; ++r) {
      float mx = acc[0][r];
      #pragma unroll
      for (int c2 = 1; c2 < 8; ++c2) mx = fmaxf(mx, acc[c2][r]);
      #pragma unroll
      for (int off = 1; off < 16; off <<= 1)   // stays within 16-lane halves
        mx = fmaxf(mx, __shfl_xor(mx, off, 32));
      const float nm = fmaxf(m_run[r], mx);
      float sm = 0.f;
      #pragma unroll
      for (int c2 = 0; c2 < 8; ++c2) sm += __expf(acc[c2][r] - nm);
      #pragma unroll
      for (int off = 1; off < 16; off <<= 1)
        sm += __shfl_xor(sm, off, 32);
      s_run[r] = s_run[r] * __expf(m_run[r] - nm) + sm;
      m_run[r] = nm;
    }
  }

  if (l16 == 0) {                       // lanes 0 and 16 hold disjoint rows
    #pragma unroll
    for (int r = 0; r < 8; ++r) {
      const int row = wrow + grp * 8 + r;
      pm[(long long)row * VSPLIT + blockIdx.y] = m_run[r];
      ps[(long long)row * VSPLIT + blockIdx.y] = s_run[r];
    }
  }
}

// ---------------- k3: merge partials + mean --------------------------------
__global__ __launch_bounds__(256)
void cce_final(const float* __restrict__ pm, const float* __restrict__ ps,
               const float* __restrict__ tl, float* __restrict__ out) {
  __shared__ float red[256];
  float local = 0.f;
  for (int row = threadIdx.x; row < NROWS; row += 256) {
    float M = -1e30f;
    #pragma unroll
    for (int i = 0; i < VSPLIT; ++i) M = fmaxf(M, pm[(long long)row * VSPLIT + i]);
    float S = 0.f;
    #pragma unroll
    for (int i = 0; i < VSPLIT; ++i)
      S += ps[(long long)row * VSPLIT + i] * __expf(pm[(long long)row * VSPLIT + i] - M);
    local += (M + __logf(S)) - tl[row];
  }
  red[threadIdx.x] = local;
  __syncthreads();
  for (int s = 128; s > 0; s >>= 1) {
    if (threadIdx.x < (unsigned)s) red[threadIdx.x] += red[threadIdx.x + s];
    __syncthreads();
  }
  if (threadIdx.x == 0) out[0] = red[0] / (float)NROWS;
}

// ---------------------------------------------------------------------------
extern "C" void kernel_launch(void* const* d_in, const int* in_sizes, int n_in,
                              void* d_out, int out_size, void* d_ws, size_t ws_size,
                              hipStream_t stream) {
  const float*     h   = (const float*)d_in[0];
  const long long* tgt = (const long long*)d_in[1];   // int64 targets
  const float*     W   = (const float*)d_in[2];
  float*           out = (float*)d_out;

  char* ws = (char*)d_ws;
  unsigned short* hb = (unsigned short*)ws;                 // 32 MiB bf16 h
  float* tl = (float*)(ws + (size_t)NROWS * DDIM * 2);      // 32 KiB
  float* pm = (float*)(ws + (size_t)NROWS * DDIM * 2 + 65536);
  float* ps = pm + (size_t)NROWS * VSPLIT;                  // 1 MiB each

  cce_convert_h<<<(NROWS * DDIM) / (256 * 8), 256, 0, stream>>>(h, hb);
  cce_target_logits<<<NROWS / 8, 256, 0, stream>>>(h, tgt, W, tl);
  dim3 grid(NROWS / MTILE, VSPLIT);   // x fast-varying -> W slice reuse in L2
  cce_lse_partial<<<grid, 256, 0, stream>>>(hb, W, pm, ps);
  cce_final<<<1, 256, 0, stream>>>(pm, ps, tl, out);
}
